// BlinkSplitCNN_5609227289127
// MI455X (gfx1250) — matrix operand order repaired and checked by reading, not compiled, not run on hardware
//
#include <hip/hip_runtime.h>

// ---------- types ----------
typedef __attribute__((ext_vector_type(16))) __bf16    v16bf;
typedef __attribute__((ext_vector_type(8)))  float     v8f;
typedef __attribute__((ext_vector_type(4)))  unsigned  v4u;
typedef __attribute__((ext_vector_type(4)))  float     v4f;

union Frag {
  v16bf v;
  unsigned short u[16];
  unsigned w32[8];
  v4u q[2];
};

// packed f32x2 -> bf16x2 (v_perm_b32 truncation; cvt_pk_bf16 builtin not available)
__device__ __forceinline__ unsigned cvt2bf(float lo, float hi) {
  union { float f; unsigned u; } a, b; a.f = lo; b.f = hi;
  return __builtin_amdgcn_perm(b.u, a.u, 0x07060302u);  // {hi[31:16], lo[31:16]}
}
__device__ __forceinline__ unsigned short f2bf(float f) {
  union { float f; unsigned u; } a; a.f = f;
  return (unsigned short)(a.u >> 16);
}

// ---------- workspace layout (bf16 elements, then f32 biases) ----------
#define OFF_CWT 0          // [64][16][128] conv weights, N-major rows of K
#define OFF_WE1 131072     // [192][192]
#define OFF_WE2 167936     // [240][192]
#define OFF_WE3 214016     // [128][256]  (n>=120, k>=240 zero)
#define OFF_WB1 246784     // [64][192]
#define OFF_WB2 259072     // [32][64]
#define OFF_WB3 261120     // [16][32]   (n>=1 zero)
#define N_BF16  261632
#define OFF_BE1 0
#define OFF_BE2 192
#define OFF_BE3 432
#define OFF_BB1 560
#define OFF_BB2 624
#define OFF_BB3 656
#define OFF_ZPAD 672       // 16 zero floats (tail-load source)
#define N_BIAS  704
#define BN_EPS  1e-3f

// ---------- prep: fold BN into bf16 transposed weights ----------
__global__ void prep_w(const float* __restrict__ conv_w,
                       const float* __restrict__ e_g1, const float* __restrict__ e_v1, const float* __restrict__ e_d1_w,
                       const float* __restrict__ e_g2, const float* __restrict__ e_v2, const float* __restrict__ e_d2_w,
                       const float* __restrict__ e_d3_w,
                       const float* __restrict__ b_g1, const float* __restrict__ b_v1, const float* __restrict__ b_d1_w,
                       const float* __restrict__ b_g2, const float* __restrict__ b_v2, const float* __restrict__ b_d2_w,
                       const float* __restrict__ b_d3_w,
                       unsigned short* __restrict__ wsb) {
  int idx = blockIdx.x * 256 + threadIdx.x;
  if (idx < OFF_WE1) {                       // conv: [w][n(16)][k(128)] = conv_w[w][k][n]
    int w = idx >> 11, r = idx & 2047;
    int n = r >> 7, k = r & 127;
    float v = (n < 3 && k < 120) ? conv_w[(w * 120 + k) * 3 + n] : 0.f;
    wsb[idx] = f2bf(v);
  } else if (idx < OFF_WE2) {                // e1: diag(s1e) @ e_d1_w, transposed
    int i = idx - OFF_WE1;
    int n = i / 192, k = i % 192;
    float s = e_g1[k] * rsqrtf(e_v1[k] + BN_EPS);
    wsb[idx] = f2bf(s * e_d1_w[k * 192 + n]);
  } else if (idx < OFF_WE3) {                // e2
    int i = idx - OFF_WE2;
    int n = i / 192, k = i % 192;
    float s = e_g2[k] * rsqrtf(e_v2[k] + BN_EPS);
    wsb[idx] = f2bf(s * e_d2_w[k * 240 + n]);
  } else if (idx < OFF_WB1) {                // e3 (padded N 120->128, K 240->256)
    int i = idx - OFF_WE3;
    int n = i >> 8, k = i & 255;
    float v = (n < 120 && k < 240) ? e_d3_w[k * 120 + n] : 0.f;
    wsb[idx] = f2bf(v);
  } else if (idx < OFF_WB2) {                // b1
    int i = idx - OFF_WB1;
    int n = i / 192, k = i % 192;
    float s = b_g1[k] * rsqrtf(b_v1[k] + BN_EPS);
    wsb[idx] = f2bf(s * b_d1_w[k * 64 + n]);
  } else if (idx < OFF_WB3) {                // b2
    int i = idx - OFF_WB2;
    int n = i >> 6, k = i & 63;
    float s = b_g2[k] * rsqrtf(b_v2[k] + BN_EPS);
    wsb[idx] = f2bf(s * b_d2_w[k * 32 + n]);
  } else if (idx < N_BF16) {                 // b3 (padded N 1->16)
    int i = idx - OFF_WB3;
    int n = i >> 5, k = i & 31;
    float v = (n == 0) ? b_d3_w[k] : 0.f;
    wsb[idx] = f2bf(v);
  }
}

__global__ void prep_b(const float* __restrict__ e_g1, const float* __restrict__ e_b1, const float* __restrict__ e_m1, const float* __restrict__ e_v1,
                       const float* __restrict__ e_d1_w, const float* __restrict__ e_d1_b,
                       const float* __restrict__ e_g2, const float* __restrict__ e_b2, const float* __restrict__ e_m2, const float* __restrict__ e_v2,
                       const float* __restrict__ e_d2_w, const float* __restrict__ e_d2_b,
                       const float* __restrict__ e_d3_b,
                       const float* __restrict__ b_g1, const float* __restrict__ b_b1, const float* __restrict__ b_m1, const float* __restrict__ b_v1,
                       const float* __restrict__ b_d1_w, const float* __restrict__ b_d1_b,
                       const float* __restrict__ b_g2, const float* __restrict__ b_b2, const float* __restrict__ b_m2, const float* __restrict__ b_v2,
                       const float* __restrict__ b_d2_w, const float* __restrict__ b_d2_b,
                       const float* __restrict__ b_d3_b,
                       float* __restrict__ fb) {
  int t = threadIdx.x;
  if (t < 192) {
    float acc = e_d1_b[t];
    for (int k = 0; k < 192; ++k) {
      float s = e_g1[k] * rsqrtf(e_v1[k] + BN_EPS);
      acc += (e_b1[k] - e_m1[k] * s) * e_d1_w[k * 192 + t];
    }
    fb[OFF_BE1 + t] = acc;
  } else if (t < 432) {
    int n = t - 192;
    float acc = e_d2_b[n];
    for (int k = 0; k < 192; ++k) {
      float s = e_g2[k] * rsqrtf(e_v2[k] + BN_EPS);
      acc += (e_b2[k] - e_m2[k] * s) * e_d2_w[k * 240 + n];
    }
    fb[OFF_BE2 + n] = acc;
  } else if (t < 560) {
    int n = t - 432;
    fb[OFF_BE3 + n] = (n < 120) ? e_d3_b[n] : 0.f;
  } else if (t < 624) {
    int n = t - 560;
    float acc = b_d1_b[n];
    for (int k = 0; k < 192; ++k) {
      float s = b_g1[k] * rsqrtf(b_v1[k] + BN_EPS);
      acc += (b_b1[k] - b_m1[k] * s) * b_d1_w[k * 64 + n];
    }
    fb[OFF_BB1 + n] = acc;
  } else if (t < 656) {
    int n = t - 624;
    float acc = b_d2_b[n];
    for (int k = 0; k < 64; ++k) {
      float s = b_g2[k] * rsqrtf(b_v2[k] + BN_EPS);
      acc += (b_b2[k] - b_m2[k] * s) * b_d2_w[k * 32 + n];
    }
    fb[OFF_BB2 + n] = acc;
  } else if (t < 672) {
    int n = t - 656;
    fb[OFF_BB3 + n] = (n == 0) ? b_d3_b[0] : 0.f;
  } else if (t < N_BIAS) {
    fb[t] = 0.f;                              // zero tail-load scratch
  }
}

// ---------- fused WMMA dense layer ----------
// IN: LDS bf16 [32][in_stride], weights: global bf16 [N][kpad] (transposed).
// This wave handles rows [m0,m0+16), output n-tiles [ntBegin, ntEnd).
template<int KT, int ACT, bool TO_LDS>
__device__ __forceinline__ void dense_layer(
    int ntBegin, int ntEnd,
    const unsigned short* __restrict__ in_lds, int in_stride,
    const unsigned short* __restrict__ wt, int kpad,
    const float* __restrict__ bias,
    unsigned short* out_lds, int out_stride,
    float* out_g, int nvalid,
    int m0, int lane) {
  const int mr = lane & 15, kh = lane >> 4;
  const unsigned short* arow = in_lds + (m0 + mr) * in_stride + 8 * kh;
  for (int nt = ntBegin; nt < ntEnd; ++nt) {
    v8f acc = {0.f, 0.f, 0.f, 0.f, 0.f, 0.f, 0.f, 0.f};
    const unsigned short* brow = wt + (nt * 16 + mr) * kpad + 8 * kh;
#pragma unroll
    for (int kt = 0; kt < KT; ++kt) {
      Frag a, b;
      a.q[0] = *(const v4u*)(arow + kt * 32);        // K = kblk+8kh .. +7
      a.q[1] = *(const v4u*)(arow + kt * 32 + 16);   // K = kblk+16+8kh .. +7
      b.q[0] = *(const v4u*)(brow + kt * 32);        // K = kblk+8kh .. +7
      b.q[1] = *(const v4u*)(brow + kt * 32 + 16);   // K = kblk+16+8kh .. +7
      acc = __builtin_amdgcn_wmma_f32_16x16x32_bf16(false, a.v, false, b.v,
                                                    (short)0, acc, false, false);
    }
    const int n = nt * 16 + mr;
    const float bn = bias[n];
#pragma unroll
    for (int j = 0; j < 8; ++j) {
      float v = acc[j] + bn;                          // D(m,n): m = j+8*kh, col = n
      if (ACT == 1) v = tanhf(v);
      else if (ACT == 2) v = 1.0f / (1.0f + __expf(-v));
      if constexpr (TO_LDS) {
        out_lds[(m0 + j + 8 * kh) * out_stride + n] = f2bf(v);
      } else {
        if (n < nvalid) out_g[(size_t)(m0 + j + 8 * kh) * 121 + n] = v;
      }
    }
  }
}

// ---------- fused main kernel ----------
// 256 threads = 8 waves. Wave: m-tile = warp&1 (rows 16*(warp&1)..), q = warp>>1 in 0..3.
// Conv: wave covers w in [16q, 16q+16) for its 16 rows (disjoint comb cols).
// Dense: the 4 waves of an m-tile split each layer by n-tile; barriers between layers.
__global__ __launch_bounds__(256) void fused_k(
    const float* __restrict__ x, const float* __restrict__ conv_b,
    const unsigned short* __restrict__ wsb, const float* __restrict__ fb,
    float* __restrict__ out) {
  __shared__ unsigned short comb[32 * 200];  // [32][192] +8 pad
  __shared__ unsigned short actA[32 * 200];
  __shared__ unsigned short actB[32 * 264];  // [32][256] +8 pad (e2 K-padded)
  __shared__ float cb_s[192];                // staged conv biases

  const int lane = threadIdx.x & 31;
  const int warp = threadIdx.x >> 5;
  const int mt   = warp & 1;
  const int q    = warp >> 1;               // 0..3
  const int m0   = mt * 16;
  const int mr   = lane & 15, kh = lane >> 4;
  const int R    = blockIdx.x * 32;

  // stage conv biases into LDS (avoids a dependent global load in the conv loop)
  if (threadIdx.x < 192) cb_s[threadIdx.x] = conv_b[threadIdx.x];

  // zero pad columns 240..255 of actB (one wave per m-tile)
  if (q == 0) {
    unsigned short* p = &actB[(m0 + mr) * 264 + 240 + 8 * kh];
#pragma unroll
    for (int i = 0; i < 8; ++i) p[i] = 0;
  }
  __syncthreads();

  // ---- conv + tanh -> comb[.][3w+k] via WMMA (A built from global fp32, NT loads) ----
  const float* xr = x + (size_t)(R + m0 + mr) * 7680 + (size_t)q * 16 * 120;
  // tail chunk (t=3, second half): kh=0 lanes read f=112..119; kh=1 lanes read zeros
  const float* tailp    = (kh == 0) ? (xr + 112) : (fb + OFF_ZPAD);
  const int    tailstep = (kh == 0) ? 120 : 0;
  for (int wi = 0; wi < 16; ++wi) {
    const int w = q * 16 + wi;
    const float* xw = xr + wi * 120;
    if (wi < 15) __builtin_prefetch(xw + 120 + 60 * kh, 0, 0);
    v8f acc = {0.f, 0.f, 0.f, 0.f, 0.f, 0.f, 0.f, 0.f};
#pragma unroll
    for (int t = 0; t < 4; ++t) {
      Frag a, b;
      const float* p1 = xw + t * 32 + 8 * kh;           // f = 32t+8kh .. +7
      v4f f0 = __builtin_nontemporal_load((const v4f*)(p1));
      v4f f1 = __builtin_nontemporal_load((const v4f*)(p1 + 4));
      v4f g0, g1;
      if (t < 3) {                                      // f = 32t+16+8kh .. +7
        g0 = __builtin_nontemporal_load((const v4f*)(p1 + 16));
        g1 = __builtin_nontemporal_load((const v4f*)(p1 + 20));
      } else {                                          // per-lane ptr, uniform control
        g0 = __builtin_nontemporal_load((const v4f*)(tailp));
        g1 = __builtin_nontemporal_load((const v4f*)(tailp + 4));
      }
      a.w32[0] = cvt2bf(f0[0], f0[1]);
      a.w32[1] = cvt2bf(f0[2], f0[3]);
      a.w32[2] = cvt2bf(f1[0], f1[1]);
      a.w32[3] = cvt2bf(f1[2], f1[3]);
      a.w32[4] = cvt2bf(g0[0], g0[1]);
      a.w32[5] = cvt2bf(g0[2], g0[3]);
      a.w32[6] = cvt2bf(g1[0], g1[1]);
      a.w32[7] = cvt2bf(g1[2], g1[3]);
      const unsigned short* br = wsb + OFF_CWT + w * 2048 + mr * 128 + t * 32 + 8 * kh;
      b.q[0] = *(const v4u*)(br);
      b.q[1] = *(const v4u*)(br + 16);
      acc = __builtin_amdgcn_wmma_f32_16x16x32_bf16(false, a.v, false, b.v,
                                                    (short)0, acc, false, false);
    }
    tailp += tailstep;
    if (mr < 3) {                                       // only N=0..2 valid
      float bn = cb_s[w * 3 + mr];
#pragma unroll
      for (int j = 0; j < 8; ++j)
        comb[(m0 + j + 8 * kh) * 200 + w * 3 + mr] = f2bf(tanhf(acc[j] + bn));
    }
  }
  __syncthreads();

  float* outR = out + (size_t)R * 121;

  // eye head
  dense_layer<6, 1, true >(3 * q, 3 * q + 3,
                           comb, 200, wsb + OFF_WE1, 192, fb + OFF_BE1, actA, 200, nullptr, 0, m0, lane);
  __syncthreads();
  dense_layer<6, 0, true >(4 * q, (q == 3) ? 15 : 4 * q + 4,
                           actA, 200, wsb + OFF_WE2, 192, fb + OFF_BE2, actB, 264, nullptr, 0, m0, lane);
  __syncthreads();
  dense_layer<8, 0, false>(2 * q, 2 * q + 2,
                           actB, 264, wsb + OFF_WE3, 256, fb + OFF_BE3, nullptr, 0, outR, 120, m0, lane);
  __syncthreads();
  // blink head
  dense_layer<6, 1, true >(q, q + 1,
                           comb, 200, wsb + OFF_WB1, 192, fb + OFF_BB1, actA, 200, nullptr, 0, m0, lane);
  __syncthreads();
  dense_layer<2, 1, true >(q, (q < 2) ? q + 1 : q,
                           actA, 200, wsb + OFF_WB2, 64, fb + OFF_BB2, actB, 264, nullptr, 0, m0, lane);
  __syncthreads();
  dense_layer<1, 2, false>(0, (q == 0) ? 1 : 0,
                           actB, 264, wsb + OFF_WB3, 32, fb + OFF_BB3, nullptr, 0, outR + 120, 1, m0, lane);
}

// ---------- launch ----------
extern "C" void kernel_launch(void* const* d_in, const int* in_sizes, int n_in,
                              void* d_out, int out_size, void* d_ws, size_t ws_size,
                              hipStream_t stream) {
  const float* x      = (const float*)d_in[0];
  const float* conv_w = (const float*)d_in[1];
  const float* conv_b = (const float*)d_in[2];
  const float* e_g1 = (const float*)d_in[3],  *e_b1 = (const float*)d_in[4];
  const float* e_m1 = (const float*)d_in[5],  *e_v1 = (const float*)d_in[6];
  const float* e_d1_w = (const float*)d_in[7], *e_d1_b = (const float*)d_in[8];
  const float* e_g2 = (const float*)d_in[9],  *e_b2 = (const float*)d_in[10];
  const float* e_m2 = (const float*)d_in[11], *e_v2 = (const float*)d_in[12];
  const float* e_d2_w = (const float*)d_in[13], *e_d2_b = (const float*)d_in[14];
  const float* e_d3_w = (const float*)d_in[15], *e_d3_b = (const float*)d_in[16];
  const float* b_g1 = (const float*)d_in[17], *b_b1 = (const float*)d_in[18];
  const float* b_m1 = (const float*)d_in[19], *b_v1 = (const float*)d_in[20];
  const float* b_d1_w = (const float*)d_in[21], *b_d1_b = (const float*)d_in[22];
  const float* b_g2 = (const float*)d_in[23], *b_b2 = (const float*)d_in[24];
  const float* b_m2 = (const float*)d_in[25], *b_v2 = (const float*)d_in[26];
  const float* b_d2_w = (const float*)d_in[27], *b_d2_b = (const float*)d_in[28];
  const float* b_d3_w = (const float*)d_in[29], *b_d3_b = (const float*)d_in[30];

  unsigned short* wsb = (unsigned short*)d_ws;
  float* fb = (float*)((char*)d_ws + (size_t)N_BF16 * 2);

  prep_w<<<(N_BF16 + 255) / 256, 256, 0, stream>>>(
      conv_w, e_g1, e_v1, e_d1_w, e_g2, e_v2, e_d2_w, e_d3_w,
      b_g1, b_v1, b_d1_w, b_g2, b_v2, b_d2_w, b_d3_w, wsb);
  prep_b<<<1, N_BIAS, 0, stream>>>(
      e_g1, e_b1, e_m1, e_v1, e_d1_w, e_d1_b,
      e_g2, e_b2, e_m2, e_v2, e_d2_w, e_d2_b, e_d3_b,
      b_g1, b_b1, b_m1, b_v1, b_d1_w, b_d1_b,
      b_g2, b_b2, b_m2, b_v2, b_d2_w, b_d2_b, b_d3_b, fb);
  fused_k<<<512, 256, 0, stream>>>(x, conv_b, wsb, fb, (float*)d_out);
}
